// ALiBiIOAttention_62380105007464
// MI455X (gfx1250) — matrix operand order repaired and checked by reading, not compiled, not run on hardware
//
#include <hip/hip_runtime.h>
#include <stdint.h>

// ---------------------------------------------------------------------------
// CDNA5 (gfx1250) ALiBi-IO attention. wave32, WMMA bf16 (f32 accumulate).
// ---------------------------------------------------------------------------

typedef __attribute__((ext_vector_type(16))) __bf16 v16bf;
typedef __attribute__((ext_vector_type(8)))  float  v8f;

union BFrag { v16bf v; uint32_t u[8]; };

#define S_LEN   2048
#define D_MODEL 1024
#define M_TOT   4096            // B*S
#define NHEAD   16
#define HDIM    64
#define IO0     1008            // D - 16

__device__ __forceinline__ uint16_t f2bf(float f) {
  uint32_t u = __float_as_uint(f);
  u += 0x7fffu + ((u >> 16) & 1u);      // round-to-nearest-even
  return (uint16_t)(u >> 16);
}

// A-fragment (16x32 bf16, M x K), row-major source with leading dim `ld`.
// ISA layout: lanes 0-15 M=lane K in {0..7,16..23}; lanes 16-31 same M,
// K in {8..15,24..31}; VGPR s holds K pair (base + 2*(s&3)).
__device__ __forceinline__ v16bf load_fragA(const uint16_t* base, int row0, int ld,
                                            int k0, int lane) {
  BFrag f;
  const int m    = row0 + (lane & 15);
  const int kadd = (lane & 16) ? 8 : 0;
#pragma unroll
  for (int s = 0; s < 8; ++s) {
    const int k = k0 + ((s < 4) ? 0 : 16) + kadd + (s & 3) * 2;
    f.u[s] = *(const uint32_t*)(base + (size_t)m * ld + k);
  }
  return f.v;
}

// B-fragment (32x16 bf16, K x N). Source stored "column-major as rows":
// baseT[n][k] contiguous in k. Lanes 0-15: K in {0..7,16..23}, lanes 16-31:
// K in {8..15,24..31}; VGPR s holds K pair (base + 2*(s&3)).
__device__ __forceinline__ v16bf load_fragB(const uint16_t* baseT, int col0, int ld,
                                            int k0, int lane) {
  BFrag f;
  const int n    = col0 + (lane & 15);
  const int kadd = (lane & 16) ? 8 : 0;
#pragma unroll
  for (int s = 0; s < 8; ++s)
    f.u[s] = *(const uint32_t*)(baseT + (size_t)n * ld + k0 + ((s < 4) ? 0 : 16) + kadd + (s & 3) * 2);
  return f.v;
}

__device__ __forceinline__ v8f wmma_bf(v16bf a, v16bf b, v8f c) {
  return __builtin_amdgcn_wmma_f32_16x16x32_bf16(false, a, false, b, (short)0, c,
                                                 false, false);
}

// ---------------------------------------------------------------------------
// prep: x -> bf16 ; wq/wk/wv/wo -> bf16 transposed ([N][K])
// ---------------------------------------------------------------------------
__global__ __launch_bounds__(256) void aiio_prep(
    const float* __restrict__ x, const float* __restrict__ wq,
    const float* __restrict__ wk, const float* __restrict__ wv,
    const float* __restrict__ wo, uint16_t* __restrict__ xbf,
    uint16_t* __restrict__ wt) {
  int idx = blockIdx.x * 256 + threadIdx.x;
  const int XN = M_TOT * D_MODEL;
  if (idx < XN) { xbf[idx] = f2bf(x[idx]); return; }
  idx -= XN;
  if (idx >= 4 * D_MODEL * D_MODEL) return;
  const int wi = idx >> 20;               // which weight matrix
  const int r  = idx & 1048575;
  const int n  = r >> 10;
  const int k  = r & 1023;
  const float* src = (wi == 0) ? wq : (wi == 1) ? wk : (wi == 2) ? wv : wo;
  wt[(size_t)wi * 1048576 + (size_t)n * 1024 + k] = f2bf(src[(size_t)k * 1024 + n]);
}

// ---------------------------------------------------------------------------
// bf16 GEMM: C[M,N] = A[M,K] * Bt[N,K]^T, bf16 out (optionally transposed C)
// One wave computes a 32x32 tile. Grid sized so no partial waves.
// ---------------------------------------------------------------------------
__global__ __launch_bounds__(256) void aiio_gemm_bf16(
    const uint16_t* __restrict__ A, const uint16_t* __restrict__ Bt,
    uint16_t* __restrict__ C, int M, int N, int K, int transposeC) {
  const int lane  = threadIdx.x & 31;
  const int wid   = (blockIdx.x * 256 + threadIdx.x) >> 5;
  const int tilesN = N >> 5;
  const int m0 = (wid / tilesN) * 32;
  const int n0 = (wid % tilesN) * 32;

  v8f acc[2][2];
#pragma unroll
  for (int a = 0; a < 2; ++a)
#pragma unroll
    for (int b = 0; b < 2; ++b) acc[a][b] = v8f{};

  for (int k0 = 0; k0 < K; k0 += 32) {
    if (k0 + 32 < K)  // pull next K-slab toward L2/L0 (global_prefetch_b8)
      __builtin_prefetch(A + (size_t)(m0 + (lane & 15)) * K + k0 + 32, 0, 1);
    v16bf a0 = load_fragA(A, m0,      K, k0, lane);
    v16bf a1 = load_fragA(A, m0 + 16, K, k0, lane);
    v16bf b0 = load_fragB(Bt, n0,      K, k0, lane);
    v16bf b1 = load_fragB(Bt, n0 + 16, K, k0, lane);
    acc[0][0] = wmma_bf(a0, b0, acc[0][0]);
    acc[0][1] = wmma_bf(a0, b1, acc[0][1]);
    acc[1][0] = wmma_bf(a1, b0, acc[1][0]);
    acc[1][1] = wmma_bf(a1, b1, acc[1][1]);
  }

  const int jn   = lane & 15;
  const int mofs = (lane & 16) ? 8 : 0;
  if (!transposeC) {
#pragma unroll
    for (int mi = 0; mi < 2; ++mi)
#pragma unroll
      for (int ni = 0; ni < 2; ++ni)
#pragma unroll
        for (int r = 0; r < 8; ++r) {
          const int m = m0 + mi * 16 + mofs + r;
          const int n = n0 + ni * 16 + jn;
          C[(size_t)m * N + n] = f2bf(acc[mi][ni][r]);
        }
  } else {
    // C^T[n][m]; consecutive VGPRs are consecutive m -> pack bf16 pairs
#pragma unroll
    for (int mi = 0; mi < 2; ++mi)
#pragma unroll
      for (int ni = 0; ni < 2; ++ni)
#pragma unroll
        for (int r = 0; r < 8; r += 2) {
          const int m = m0 + mi * 16 + mofs + r;  // even
          const int n = n0 + ni * 16 + jn;
          uint32_t pk = (uint32_t)f2bf(acc[mi][ni][r]) |
                        ((uint32_t)f2bf(acc[mi][ni][r + 1]) << 16);
          *(uint32_t*)(C + (size_t)n * M + m) = pk;
        }
  }
}

// ---------------------------------------------------------------------------
// attention: one wave per (b, h, 16-query tile). Flash-style online softmax.
// qbf/kbf row-major [4096][1024]; vt transposed [1024][4096].
// ---------------------------------------------------------------------------
__global__ __launch_bounds__(256) void aiio_attn(
    const uint16_t* __restrict__ qbf, const uint16_t* __restrict__ kbf,
    const uint16_t* __restrict__ vt, const float* __restrict__ x,
    const float* __restrict__ is_get, uint16_t* __restrict__ attn_out) {
  __shared__ uint16_t ldsp[8][16 * 32];   // per-wave P staging tile (bf16)

  const int lane = threadIdx.x & 31;
  const int w    = threadIdx.x >> 5;
  const int wid  = blockIdx.x * 8 + w;
  const int qt = wid & 127;
  const int h  = (wid >> 7) & 15;
  const int b  = wid >> 11;
  const int i0 = qt * 16;
  const int rowq = b * S_LEN + i0;
  const int jn   = lane & 15;
  const int mofs = (lane & 16) ? 8 : 0;

  const v16bf qA0 = load_fragA(qbf, rowq, D_MODEL, h * HDIM,      lane);
  const v16bf qA1 = load_fragA(qbf, rowq, D_MODEL, h * HDIM + 32, lane);

  v8f acc[4];
#pragma unroll
  for (int c = 0; c < 4; ++c) acc[c] = v8f{};
  float mrow[8], lrow[8];
#pragma unroll
  for (int r = 0; r < 8; ++r) { mrow[r] = -3.0e38f; lrow[r] = 0.0f; }

  const float slope = exp2f(-0.5f * (float)(h + 1));   // ratio^(h+1), ratio=2^-0.5

  float tgt[8]; int getc[8];
#pragma unroll
  for (int r = 0; r < 8; ++r) { tgt[r] = 0.0f; getc[r] = 0; }
  if (h == 0) {
#pragma unroll
    for (int r = 0; r < 8; ++r) {
      const int gi = b * S_LEN + i0 + mofs + r;
      tgt[r]  = x[(size_t)gi * D_MODEL + IO0] + 1.0f + x[(size_t)gi * D_MODEL + IO0 + 1];
      getc[r] = (is_get[gi] > 0.5f) ? 1 : 0;
    }
  }

  const int nk2 = (h == 0) ? (S_LEN / 32) : ((i0 + 15) / 32 + 1);

  for (int kt = 0; kt < nk2; ++kt) {
    const int j0 = kt * 32;

    // ---- scores: two 16x16 tiles (32 keys), K=64 contraction ----
    float sa[2][8];
#pragma unroll
    for (int t = 0; t < 2; ++t) {
      v16bf kb0 = load_fragB(kbf, b * S_LEN + j0 + t * 16, D_MODEL, h * HDIM,      lane);
      v16bf kb1 = load_fragB(kbf, b * S_LEN + j0 + t * 16, D_MODEL, h * HDIM + 32, lane);
      v8f s = v8f{};
      s = wmma_bf(qA0, kb0, s);
      s = wmma_bf(qA1, kb1, s);
#pragma unroll
      for (int r = 0; r < 8; ++r) {
        const int i = i0 + mofs + r;
        const int j = j0 + t * 16 + jn;
        float val = s[r] * 0.125f - slope * (float)(i - j);
        if (j > i) val -= 1.0e9f;                         // causal
        if (h == 0 && getc[r])
          val = -slope * fabsf(tgt[r] - (float)j);        // read-head bias
        sa[t][r] = val;
      }
    }

    // ---- online softmax update (row = 16 lanes of one half-wave) ----
    float p[2][8];
#pragma unroll
    for (int r = 0; r < 8; ++r) {
      float mx = fmaxf(sa[0][r], sa[1][r]);
#pragma unroll
      for (int o = 1; o < 16; o <<= 1) mx = fmaxf(mx, __shfl_xor(mx, o, 32));
      const float mnew  = fmaxf(mrow[r], mx);
      const float alpha = __expf(mrow[r] - mnew);
      float sum = 0.0f;
#pragma unroll
      for (int t = 0; t < 2; ++t) { p[t][r] = __expf(sa[t][r] - mnew); sum += p[t][r]; }
#pragma unroll
      for (int o = 1; o < 16; o <<= 1) sum += __shfl_xor(sum, o, 32);
      lrow[r] = lrow[r] * alpha + sum;
      mrow[r] = mnew;
#pragma unroll
      for (int c = 0; c < 4; ++c) acc[c][r] *= alpha;
    }

    // ---- transpose P (C-layout -> A-layout) through per-wave LDS tile ----
#pragma unroll
    for (int t = 0; t < 2; ++t)
#pragma unroll
      for (int r = 0; r < 8; ++r)
        ldsp[w][(mofs + r) * 32 + t * 16 + jn] = f2bf(p[t][r]);
    asm volatile("s_wait_dscnt 0x0" ::: "memory");   // DS in-order per wave
    BFrag pf;
    {
      const int mA   = lane & 15;
      const int kadd = (lane & 16) ? 8 : 0;
#pragma unroll
      for (int s = 0; s < 8; ++s) {
        const int k = ((s < 4) ? 0 : 16) + kadd + (s & 3) * 2;
        pf.u[s] = *(const uint32_t*)(&ldsp[w][mA * 32 + k]);
      }
    }
    asm volatile("s_wait_dscnt 0x0" ::: "memory");

    // ---- acc += P(16x32) @ V(32x64) ----
#pragma unroll
    for (int c = 0; c < 4; ++c) {
      v16bf vb = load_fragB(vt, h * HDIM + c * 16, M_TOT, b * S_LEN + j0, lane);
      acc[c] = wmma_bf(pf.v, vb, acc[c]);
    }
  }

  // ---- normalize + store bf16 ----
#pragma unroll
  for (int c = 0; c < 4; ++c)
#pragma unroll
    for (int r = 0; r < 8; ++r) {
      const int i = i0 + mofs + r;
      const float o = acc[c][r] / lrow[r];
      attn_out[(size_t)(b * S_LEN + i) * D_MODEL + h * HDIM + c * 16 + jn] = f2bf(o);
    }
}

// ---------------------------------------------------------------------------
// final: out = x + attn_out @ wo, plus I/O channel updates. fp32 out.
// ---------------------------------------------------------------------------
__global__ __launch_bounds__(256) void aiio_gemm_final(
    const uint16_t* __restrict__ A, const uint16_t* __restrict__ Bt,
    const float* __restrict__ x, const float* __restrict__ is_get,
    const float* __restrict__ is_put, float* __restrict__ out,
    int M, int N, int K) {
  const int lane = threadIdx.x & 31;
  const int wid  = (blockIdx.x * 256 + threadIdx.x) >> 5;
  const int tilesN = N >> 5;
  const int m0 = (wid / tilesN) * 32;
  const int n0 = (wid % tilesN) * 32;

  v8f acc[2][2];
#pragma unroll
  for (int a = 0; a < 2; ++a)
#pragma unroll
    for (int b = 0; b < 2; ++b) acc[a][b] = v8f{};

  for (int k0 = 0; k0 < K; k0 += 32) {
    v16bf a0 = load_fragA(A, m0,      K, k0, lane);
    v16bf a1 = load_fragA(A, m0 + 16, K, k0, lane);
    v16bf b0 = load_fragB(Bt, n0,      K, k0, lane);
    v16bf b1 = load_fragB(Bt, n0 + 16, K, k0, lane);
    acc[0][0] = wmma_bf(a0, b0, acc[0][0]);
    acc[0][1] = wmma_bf(a0, b1, acc[0][1]);
    acc[1][0] = wmma_bf(a1, b0, acc[1][0]);
    acc[1][1] = wmma_bf(a1, b1, acc[1][1]);
  }

  const int jn   = lane & 15;
  const int mofs = (lane & 16) ? 8 : 0;
#pragma unroll
  for (int mi = 0; mi < 2; ++mi)
#pragma unroll
    for (int ni = 0; ni < 2; ++ni)
#pragma unroll
      for (int r = 0; r < 8; ++r) {
        const int m = m0 + mi * 16 + mofs + r;
        const int n = n0 + ni * 16 + jn;
        float o = x[(size_t)m * N + n] + acc[mi][ni][r];
        if (n == IO0 + 1)       o += is_get[m];   // read_offset += getchar
        else if (n == IO0 + 3)  o += is_put[m];   // write_offset += putchar
        else if (n == IO0 + 13) o  = is_put[m];   // output_ready = putchar
        out[(size_t)m * N + n] = o;
      }
}

// ---------------------------------------------------------------------------
extern "C" void kernel_launch(void* const* d_in, const int* in_sizes, int n_in,
                              void* d_out, int out_size, void* d_ws, size_t ws_size,
                              hipStream_t stream) {
  const float* x      = (const float*)d_in[0];
  const float* is_get = (const float*)d_in[1];
  const float* is_put = (const float*)d_in[2];
  const float* wq     = (const float*)d_in[3];
  const float* wk     = (const float*)d_in[4];
  const float* wv     = (const float*)d_in[5];
  const float* wo     = (const float*)d_in[6];

  // workspace layout (bf16 elements): 6 x 4M = 48 MiB
  uint16_t* xbf  = (uint16_t*)d_ws;
  uint16_t* wt   = xbf  + (size_t)M_TOT * D_MODEL;     // 4 x [1024][1024] transposed
  uint16_t* qbf  = wt   + (size_t)4 * D_MODEL * D_MODEL;
  uint16_t* kbf  = qbf  + (size_t)M_TOT * D_MODEL;
  uint16_t* vt   = kbf  + (size_t)M_TOT * D_MODEL;     // transposed [1024][4096]
  uint16_t* aobf = vt   + (size_t)M_TOT * D_MODEL;

  // 1) convert / transpose inputs to bf16
  aiio_prep<<<(M_TOT * D_MODEL + 4 * D_MODEL * D_MODEL) / 256, 256, 0, stream>>>(
      x, wq, wk, wv, wo, xbf, wt);

  // 2) QKV projections (each: 128x32 tiles = 4096 waves = 512 blocks)
  aiio_gemm_bf16<<<512, 256, 0, stream>>>(xbf, wt,               qbf, M_TOT, D_MODEL, D_MODEL, 0);
  aiio_gemm_bf16<<<512, 256, 0, stream>>>(xbf, wt + 1 * 1048576, kbf, M_TOT, D_MODEL, D_MODEL, 0);
  aiio_gemm_bf16<<<512, 256, 0, stream>>>(xbf, wt + 2 * 1048576, vt,  M_TOT, D_MODEL, D_MODEL, 1);

  // 3) attention: 2*16*128 = 4096 waves = 512 blocks
  aiio_attn<<<512, 256, 0, stream>>>(qbf, kbf, vt, x, is_get, aobf);

  // 4) output projection + residual + I/O epilogue
  aiio_gemm_final<<<512, 256, 0, stream>>>(aobf, wt + 3 * 1048576, x, is_get, is_put,
                                           (float*)d_out, M_TOT, D_MODEL, D_MODEL);
}
